// NunchakuQwenImageTransformerBlock_11338713661524
// MI455X (gfx1250) — matrix operand order repaired and checked by reading, not compiled, not run on hardware
//
#include <hip/hip_runtime.h>
#include <hip/hip_bf16.h>
#include <cstdint>

typedef __attribute__((ext_vector_type(16))) _Float16 v16h;
typedef __attribute__((ext_vector_type(8)))  float    v8f;

union FragU { v16h v; uint4 u[2]; };

// A-fragment (16x32 f16): VGPR0-3 hold K = half*8+0..7 (pairs), VGPR4-7 hold K = 16+half*8+0..7
__device__ __forceinline__ v16h frag_a(const _Float16* rowbase) {
    const int half = (threadIdx.x >> 4) & 1;
    FragU f;
    f.u[0] = *(const uint4*)(rowbase + half * 8);
    f.u[1] = *(const uint4*)(rowbase + 16 + half * 8);
    return f.v;
}
// B-fragment (32x16 f16, stored [N][K] K-contiguous): VGPR0-3 hold K = half*8+0..7, VGPR4-7 hold K = 16+half*8+0..7
__device__ __forceinline__ v16h frag_b(const _Float16* rowbase) {
    const int half = (threadIdx.x >> 4) & 1;
    FragU f;
    f.u[0] = *(const uint4*)(rowbase + half * 8);
    f.u[1] = *(const uint4*)(rowbase + 16 + half * 8);
    return f.v;
}

// CDNA5 async global->LDS copy (16B per lane), tracked by ASYNCcnt.
// LDS byte address = low 32 bits of the generic pointer (ISA 10.2).
__device__ __forceinline__ void async_b128(const void* g, void* l) {
    uint32_t laddr = (uint32_t)(uintptr_t)l;
    asm volatile("global_load_async_to_lds_b128 %0, %1, off"
                 :: "v"(laddr), "v"(g) : "memory");
}
__device__ __forceinline__ void wait_async0() {
    asm volatile("s_wait_asynccnt 0x0" ::: "memory");
}

__device__ __forceinline__ float gelu_t(float x) {
    float x3 = x * x * x;
    return 0.5f * x * (1.f + tanhf(0.7978845608028654f * (x + 0.044715f * x3)));
}

// ---------------- prep: silu(temb), key-mask bias vector ----------------
__global__ __launch_bounds__(256) void prep_kernel(const float* __restrict__ temb,
                                                   const float* __restrict__ mask,
                                                   float* __restrict__ silu_t,
                                                   float* __restrict__ biasv) {
    int i = blockIdx.x * 256 + threadIdx.x;
    if (i < 3072) { float t = temb[i]; silu_t[i] = t / (1.f + __expf(-t)); }
    if (i < 1280) { biasv[i] = (i < 256) ? (mask[i] > 0.f ? 0.f : -1e9f) : 0.f; }
}

// ---------------- modulation params: one wave per output, compact [j][D] layout ----------------
__global__ __launch_bounds__(256) void modparams_kernel(
    const float* __restrict__ silu_t,
    const float* __restrict__ wImg, const float* __restrict__ bImg,
    const float* __restrict__ wTxt, const float* __restrict__ bTxt,
    float* __restrict__ outImg, float* __restrict__ outTxt) {
    const int lane = threadIdx.x & 31;
    const int gw = blockIdx.x * 8 + (threadIdx.x >> 5);   // 2*6*3072 waves
    const int set = gw / 18432;
    const int oo  = gw % 18432;
    const int j = oo / 3072, d = oo % 3072;
    const int wrow = d * 6 + j;                            // m.reshape(D,6) index
    const float* wp = (set ? wTxt : wImg) + (size_t)wrow * 3072;
    float s = 0.f;
    for (int k = lane; k < 3072; k += 32) s += silu_t[k] * wp[k];
    for (int o = 16; o > 0; o >>= 1) s += __shfl_down(s, o);
    if (lane == 0) {
        float* dst = set ? outTxt : outImg;
        const float* bp = set ? bTxt : bImg;
        dst[j * 3072 + d] = s + bp[wrow];
    }
}

// ---------------- LayerNorm + modulate -> f16 ----------------
__global__ __launch_bounds__(256) void ln_mod_kernel(
    const float* __restrict__ x, const float* __restrict__ shiftp,
    const float* __restrict__ scalep, _Float16* __restrict__ out) {
    __shared__ float2 red[8];
    const int tid = threadIdx.x;
    const size_t row = blockIdx.x;
    const float* xr = x + row * 3072;
    float vals[12];
    float s = 0.f, ss = 0.f;
#pragma unroll
    for (int i = 0; i < 12; ++i) { float v = xr[tid + i * 256]; vals[i] = v; s += v; ss += v * v; }
    int lane = tid & 31, w = tid >> 5;
    for (int o = 16; o > 0; o >>= 1) { s += __shfl_down(s, o); ss += __shfl_down(ss, o); }
    if (lane == 0) red[w] = make_float2(s, ss);
    __syncthreads();
    if (tid == 0) {
        float a = 0.f, b = 0.f;
        for (int i = 0; i < 8; i++) { a += red[i].x; b += red[i].y; }
        red[0] = make_float2(a, b);
    }
    __syncthreads();
    const float mu = red[0].x * (1.f / 3072.f);
    const float var = red[0].y * (1.f / 3072.f) - mu * mu;
    const float rstd = rsqrtf(var + 1e-6f);
    _Float16* orow = out + row * 3072;
#pragma unroll
    for (int i = 0; i < 12; ++i) {
        int d = tid + i * 256;
        orow[d] = (_Float16)((vals[i] - mu) * rstd * (scalep[d] + 1.f) + shiftp[d]);
    }
}

// ---------------- WMMA GEMM: C = A(f16,[M][K]) @ W^T(f32,[N][K]) + bias ----------------
// 128x128 block tile, BK=32, double-buffered: A tile via async global->LDS copy,
// B tile register-staged with f32->f16 conversion.
// EPI 0: bias -> f32 out; EPI 1: bias+gelu -> f16 out; EPI 2: residual + gate[n]*(v+bias) -> f32 out
template <int EPI>
__global__ __launch_bounds__(256) void gemm_kernel(
    const _Float16* __restrict__ A, const float* __restrict__ W,
    const float* __restrict__ bias, void* __restrict__ outp,
    const float* __restrict__ residual, const float* __restrict__ gate,
    int M, int N, int K) {
    __shared__ _Float16 As[2][128 * 40];
    __shared__ _Float16 Bs[2][128 * 40];
    const int tid = threadIdx.x;
    const int lane = tid & 31;
    const int wave = tid >> 5;
    const int wm = wave >> 1, wn = wave & 1;       // 4x2 wave grid, 32(M)x64(N) per wave
    const int bm = blockIdx.y * 128, bn = blockIdx.x * 128;
    const int l15 = lane & 15;
    (void)M;

    v8f acc[2][4] = {};
    float4 bR[4];

    auto asyncA = [&](int buf, int k0) {           // A tile: 128x32 f16 = 512 x 16B, async
#pragma unroll
        for (int t = 0; t < 2; t++) {
            int i = tid + t * 256, row = i >> 2, c = i & 3;
            async_b128(A + (size_t)(bm + row) * K + k0 + c * 8, &As[buf][row * 40 + c * 8]);
        }
    };
    auto gloadB = [&](int k0) {                    // B tile: 128x32 f32 = 1024 float4
#pragma unroll
        for (int t = 0; t < 4; t++) {
            int i = tid + t * 256, row = i >> 3, c = i & 7;
            bR[t] = *(const float4*)(W + (size_t)(bn + row) * K + k0 + c * 4);
        }
    };
    auto lstoreB = [&](int buf) {
#pragma unroll
        for (int t = 0; t < 4; t++) {
            int i = tid + t * 256, row = i >> 3, c = i & 7;
            _Float16* dst = &Bs[buf][row * 40 + c * 4];
            dst[0] = (_Float16)bR[t].x; dst[1] = (_Float16)bR[t].y;
            dst[2] = (_Float16)bR[t].z; dst[3] = (_Float16)bR[t].w;
        }
    };

    asyncA(0, 0);
    gloadB(0);
    lstoreB(0);
    wait_async0();
    __syncthreads();
    const int nk = K >> 5;
    for (int kt = 0; kt < nk; kt++) {
        const int cur = kt & 1;
        const bool more = (kt + 1 < nk);
        if (more) {
            asyncA(cur ^ 1, (kt + 1) << 5);        // next A tile streams into LDS in background
            gloadB((kt + 1) << 5);                 // next B tile global loads in flight
        }
        v16h af[2], bf[4];
#pragma unroll
        for (int mf = 0; mf < 2; mf++) af[mf] = frag_a(&As[cur][(wm * 32 + mf * 16 + l15) * 40]);
#pragma unroll
        for (int nf = 0; nf < 4; nf++) bf[nf] = frag_b(&Bs[cur][(wn * 64 + nf * 16 + l15) * 40]);
#pragma unroll
        for (int mf = 0; mf < 2; mf++)
#pragma unroll
            for (int nf = 0; nf < 4; nf++)
                acc[mf][nf] = __builtin_amdgcn_wmma_f32_16x16x32_f16(
                    false, af[mf], false, bf[nf], (short)0, acc[mf][nf], false, false);
        if (more) lstoreB(cur ^ 1);                // convert+store after compute issued
        wait_async0();
        __syncthreads();
    }

    const int halfm = (lane >> 4) & 1;
#pragma unroll
    for (int mf = 0; mf < 2; mf++) {
#pragma unroll
        for (int nf = 0; nf < 4; nf++) {
            const int col = bn + wn * 64 + nf * 16 + l15;
            const float bv = bias[col];
            float gv = 0.f;
            if (EPI == 2) gv = gate[col];
#pragma unroll
            for (int r = 0; r < 8; r++) {
                const int row = bm + wm * 32 + mf * 16 + r + halfm * 8;
                float v = acc[mf][nf][r] + bv;
                if (EPI == 0) {
                    ((float*)outp)[(size_t)row * N + col] = v;
                } else if (EPI == 1) {
                    ((_Float16*)outp)[(size_t)row * N + col] = (_Float16)gelu_t(v);
                } else {
                    ((float*)outp)[(size_t)row * N + col] = residual[(size_t)row * N + col] + gv * v;
                }
            }
        }
    }
}

// ---------------- RMSNorm(q,k) + RoPE -> f16 q/k/v, head-major [H][S][HD] ----------------
__global__ __launch_bounds__(128) void qk_rope_kernel(
    const float* __restrict__ qkvTxt, const float* __restrict__ qkvImg,
    const float* __restrict__ nqw, const float* __restrict__ nkw,
    const float* __restrict__ naqw, const float* __restrict__ nakw,
    const float* __restrict__ cosT, const float* __restrict__ sinT,
    _Float16* __restrict__ qf, _Float16* __restrict__ kf, _Float16* __restrict__ vf) {
    const int sh = blockIdx.x;
    const int h = sh % 24;
    const int s = sh / 24;
    const float* src; const float* qw; const float* kw;
    if (s < 256) { src = qkvTxt + (size_t)s * 9216; qw = naqw; kw = nakw; }
    else         { src = qkvImg + (size_t)(s - 256) * 9216; qw = nqw; kw = nkw; }
    const int d = threadIdx.x;                 // 0..127
    float q = src[h * 128 + d];
    float k = src[3072 + h * 128 + d];
    float v = src[6144 + h * 128 + d];
    __shared__ float2 red[4];
    float a = q * q, b = k * k;
    int lane = d & 31, w = d >> 5;
    for (int o = 16; o > 0; o >>= 1) { a += __shfl_down(a, o); b += __shfl_down(b, o); }
    if (lane == 0) red[w] = make_float2(a, b);
    __syncthreads();
    float qs = red[0].x + red[1].x + red[2].x + red[3].x;
    float ks = red[0].y + red[1].y + red[2].y + red[3].y;
    float qn = q * rsqrtf(qs * (1.f / 128.f) + 1e-6f) * qw[d];
    float kn = k * rsqrtf(ks * (1.f / 128.f) + 1e-6f) * kw[d];
    const int i = d >> 1;
    float c = cosT[s * 64 + i], sn = sinT[s * 64 + i];
    float qo_o = __shfl_xor(qn, 1);
    float ko_o = __shfl_xor(kn, 1);
    float qo = (d & 1) ? (qo_o * sn + qn * c) : (qn * c - qo_o * sn);
    float ko = (d & 1) ? (ko_o * sn + kn * c) : (kn * c - ko_o * sn);
    const size_t idx = ((size_t)h * 1280 + s) * 128 + d;
    qf[idx] = (_Float16)qo;
    kf[idx] = (_Float16)ko;
    vf[idx] = (_Float16)v;
}

// ---------------- flash attention: one (head, 128-query block) per workgroup ----------------
#define KBLK 64
#define KS_LD 136
#define VT_LD 72
#define PL_LD 72
__global__ __launch_bounds__(256) void attn_kernel(
    const _Float16* __restrict__ qg, const _Float16* __restrict__ kg,
    const _Float16* __restrict__ vg, const float* __restrict__ biasv,
    _Float16* __restrict__ outp) {
    __shared__ _Float16 Ks[KBLK * KS_LD];
    __shared__ _Float16 Vt[128 * VT_LD];
    __shared__ _Float16 Pl[128 * PL_LD];
    __shared__ float mS[128], lS[128], aS[128];
    __shared__ float pmax[256], psum[256];

    const int tid = threadIdx.x;
    const int lane = tid & 31;
    const int wave = tid >> 5;
    const int wm = wave >> 1;      // q-tile (4 x 32 rows)
    const int wn = wave & 1;       // key-slice (scores) / hd-slice (O)
    const int half = (lane >> 4) & 1;
    const int l15 = lane & 15;
    const int h = blockIdx.x;
    const int qb = blockIdx.y;
    const _Float16* qh = qg + ((size_t)h * 1280 + qb * 128) * 128;

    // Q fragments resident in registers for the whole block
    v16h qa[2][4];
#pragma unroll
    for (int mf = 0; mf < 2; mf++) {
        const _Float16* rb = qh + (size_t)(wm * 32 + mf * 16 + l15) * 128;
#pragma unroll
        for (int kk = 0; kk < 4; kk++) qa[mf][kk] = frag_a(rb + kk * 32);
    }
    if (tid < 128) { mS[tid] = -3.0e38f; lS[tid] = 0.f; }
    v8f o[2][4] = {};
    __syncthreads();

    const float sc = 0.08838834764831845f;   // 1/sqrt(128)
    for (int kb = 0; kb < 1280; kb += KBLK) {
        const _Float16* kh = kg + ((size_t)h * 1280 + kb) * 128;
        const _Float16* vh = vg + ((size_t)h * 1280 + kb) * 128;
        for (int i = tid; i < KBLK * 16; i += 256) {       // K tile: async global->LDS
            int row = i >> 4, c = i & 15;
            async_b128(kh + (size_t)row * 128 + c * 8, &Ks[row * KS_LD + c * 8]);
        }
        for (int i = tid; i < KBLK * 128; i += 256) {      // V^T tile: element transpose
            int key = i >> 7, d = i & 127;
            Vt[d * VT_LD + key] = vh[(size_t)key * 128 + d];
        }
        wait_async0();
        __syncthreads();

        // scores S = Q K^T for this wave's 32q x 32k slice
        v8f s[2][2] = {};
#pragma unroll
        for (int kk = 0; kk < 4; kk++) {
            v16h bf[2];
#pragma unroll
            for (int nf = 0; nf < 2; nf++)
                bf[nf] = frag_b(&Ks[(wn * 32 + nf * 16 + l15) * KS_LD + kk * 32]);
#pragma unroll
            for (int mf = 0; mf < 2; mf++)
#pragma unroll
                for (int nf = 0; nf < 2; nf++)
                    s[mf][nf] = __builtin_amdgcn_wmma_f32_16x16x32_f16(
                        false, qa[mf][kk], false, bf[nf], (short)0, s[mf][nf], false, false);
        }
        float bvv[2];
#pragma unroll
        for (int nf = 0; nf < 2; nf++) bvv[nf] = biasv[kb + wn * 32 + nf * 16 + l15];
#pragma unroll
        for (int mf = 0; mf < 2; mf++)
#pragma unroll
            for (int nf = 0; nf < 2; nf++)
#pragma unroll
                for (int r = 0; r < 8; r++)
                    s[mf][nf][r] = s[mf][nf][r] * sc + bvv[nf];

        // per-row max within this wave's key slice (16-lane groups share a row)
#pragma unroll
        for (int mf = 0; mf < 2; mf++)
#pragma unroll
            for (int r = 0; r < 8; r++) {
                float mw = fmaxf(s[mf][0][r], s[mf][1][r]);
#pragma unroll
                for (int off = 1; off < 16; off <<= 1) mw = fmaxf(mw, __shfl_xor(mw, off));
                if (l15 == 0) pmax[(wm * 32 + mf * 16 + r + half * 8) * 2 + wn] = mw;
            }
        __syncthreads();

        float rowm[16];
#pragma unroll
        for (int mf = 0; mf < 2; mf++)
#pragma unroll
            for (int r = 0; r < 8; r++) {
                const int row = wm * 32 + mf * 16 + r + half * 8;
                float mNew = fmaxf(mS[row], fmaxf(pmax[row * 2], pmax[row * 2 + 1]));
                rowm[mf * 8 + r] = mNew;
                float p0 = __expf(s[mf][0][r] - mNew);
                float p1 = __expf(s[mf][1][r] - mNew);
                Pl[row * PL_LD + wn * 32 + l15]      = (_Float16)p0;
                Pl[row * PL_LD + wn * 32 + 16 + l15] = (_Float16)p1;
                float sw = p0 + p1;
#pragma unroll
                for (int off = 1; off < 16; off <<= 1) sw += __shfl_xor(sw, off);
                if (l15 == 0) psum[row * 2 + wn] = sw;
            }
        __syncthreads();
        if (wn == 0 && l15 == 0) {
#pragma unroll
            for (int mf = 0; mf < 2; mf++)
#pragma unroll
                for (int r = 0; r < 8; r++) {
                    const int row = wm * 32 + mf * 16 + r + half * 8;
                    const float mNew = rowm[mf * 8 + r];
                    const float al = __expf(mS[row] - mNew);
                    lS[row] = al * lS[row] + psum[row * 2] + psum[row * 2 + 1];
                    mS[row] = mNew;
                    aS[row] = al;
                }
        }
        __syncthreads();

        // rescale O and accumulate O += P @ V^T (wave covers 32q x 64hd)
#pragma unroll
        for (int mf = 0; mf < 2; mf++)
#pragma unroll
            for (int r = 0; r < 8; r++) {
                const float al = aS[wm * 32 + mf * 16 + r + half * 8];
#pragma unroll
                for (int nf = 0; nf < 4; nf++) o[mf][nf][r] *= al;
            }
#pragma unroll
        for (int kk = 0; kk < 2; kk++) {
            v16h pa[2], vb[4];
#pragma unroll
            for (int mf = 0; mf < 2; mf++)
                pa[mf] = frag_a(&Pl[(wm * 32 + mf * 16 + l15) * PL_LD + kk * 32]);
#pragma unroll
            for (int nf = 0; nf < 4; nf++)
                vb[nf] = frag_b(&Vt[(wn * 64 + nf * 16 + l15) * VT_LD + kk * 32]);
#pragma unroll
            for (int mf = 0; mf < 2; mf++)
#pragma unroll
                for (int nf = 0; nf < 4; nf++)
                    o[mf][nf] = __builtin_amdgcn_wmma_f32_16x16x32_f16(
                        false, pa[mf], false, vb[nf], (short)0, o[mf][nf], false, false);
        }
        __syncthreads();
    }

    // write normalized output as f16, layout [s][h*128+hd]
#pragma unroll
    for (int mf = 0; mf < 2; mf++)
#pragma unroll
        for (int r = 0; r < 8; r++) {
            const int rl = wm * 32 + mf * 16 + r + half * 8;
            const float inv = 1.f / lS[rl];
            const size_t sg = (size_t)(qb * 128 + rl);
#pragma unroll
            for (int nf = 0; nf < 4; nf++) {
                const int dh = wn * 64 + nf * 16 + l15;
                outp[sg * 3072 + h * 128 + dh] = (_Float16)(o[mf][nf][r] * inv);
            }
        }
}

// ---------------- host launch ----------------
extern "C" void kernel_launch(void* const* d_in, const int* in_sizes, int n_in,
                              void* d_out, int out_size, void* d_ws, size_t ws_size,
                              hipStream_t stream) {
    (void)in_sizes; (void)n_in; (void)out_size;
    const float* hidden = (const float*)d_in[0];
    const float* enc    = (const float*)d_in[1];
    const float* mask   = (const float*)d_in[2];
    const float* temb   = (const float*)d_in[3];
    const float* ropeC  = (const float*)d_in[4];
    const float* ropeS  = (const float*)d_in[5];
    const float* imw = (const float*)d_in[6];
    const float* imb = (const float*)d_in[7];
    const float* tmw = (const float*)d_in[8];
    const float* tmb = (const float*)d_in[9];
    const float* qkvw  = (const float*)d_in[10];
    const float* qkvb  = (const float*)d_in[11];
    const float* aqkvw = (const float*)d_in[12];
    const float* aqkvb = (const float*)d_in[13];
    const float* nqw  = (const float*)d_in[14];
    const float* nkw  = (const float*)d_in[15];
    const float* naqw = (const float*)d_in[16];
    const float* nakw = (const float*)d_in[17];
    const float* tow  = (const float*)d_in[18];
    const float* tob  = (const float*)d_in[19];
    const float* taow = (const float*)d_in[20];
    const float* taob = (const float*)d_in[21];
    const float* iw1 = (const float*)d_in[22];
    const float* ib1 = (const float*)d_in[23];
    const float* iw2 = (const float*)d_in[24];
    const float* ib2 = (const float*)d_in[25];
    const float* tw1 = (const float*)d_in[26];
    const float* tb1 = (const float*)d_in[27];
    const float* tw2 = (const float*)d_in[28];
    const float* tb2 = (const float*)d_in[29];
    float* outF = (float*)d_out;

    char* base = (char*)d_ws;
    size_t off = 0;
    auto take = [&](size_t bytes) -> void* {
        void* p = base + off;
        off = (off + bytes + 255) & ~(size_t)255;
        return p;
    };
    float* silu   = (float*)take(3072 * 4);
    float* biasv  = (float*)take(1280 * 4);
    float* modImg = (float*)take(6 * 3072 * 4);           // [shift1,scale1,gate1,shift2,scale2,gate2][D]
    float* modTxt = (float*)take(6 * 3072 * 4);
    _Float16* xImg = (_Float16*)take((size_t)1024 * 3072 * 2);   // ln1-modulated, reused for ln2
    _Float16* xTxt = (_Float16*)take((size_t)256 * 3072 * 2);
    _Float16* qF = (_Float16*)take((size_t)24 * 1280 * 128 * 2); // head-major
    _Float16* kF = (_Float16*)take((size_t)24 * 1280 * 128 * 2);
    _Float16* vF = (_Float16*)take((size_t)24 * 1280 * 128 * 2);
    _Float16* attnF = (_Float16*)take((size_t)1280 * 3072 * 2);
    float* hImg = (float*)take((size_t)1024 * 3072 * 4);
    float* hTxt = (float*)take((size_t)256 * 3072 * 4);
    float* qkvImg = (float*)take((size_t)1024 * 9216 * 4);
    float* qkvTxt = (float*)take((size_t)256 * 9216 * 4);
    _Float16* midImg = (_Float16*)(void*)qkvImg;   // reuse qkv region after rope (25.2MB <= 37.7MB)
    _Float16* midTxt = (_Float16*)(void*)qkvTxt;   // (6.3MB <= 9.4MB)
    if (off > ws_size) return;

    prep_kernel<<<12, 256, 0, stream>>>(temb, mask, silu, biasv);
    modparams_kernel<<<4608, 256, 0, stream>>>(silu, imw, imb, tmw, tmb, modImg, modTxt);
    ln_mod_kernel<<<1024, 256, 0, stream>>>(hidden, modImg, modImg + 3072, xImg);
    ln_mod_kernel<<<256, 256, 0, stream>>>(enc, modTxt, modTxt + 3072, xTxt);
    gemm_kernel<0><<<dim3(72, 8), 256, 0, stream>>>(xImg, qkvw, qkvb, qkvImg, nullptr, nullptr, 1024, 9216, 3072);
    gemm_kernel<0><<<dim3(72, 2), 256, 0, stream>>>(xTxt, aqkvw, aqkvb, qkvTxt, nullptr, nullptr, 256, 9216, 3072);
    qk_rope_kernel<<<30720, 128, 0, stream>>>(qkvTxt, qkvImg, nqw, nkw, naqw, nakw, ropeC, ropeS, qF, kF, vF);
    attn_kernel<<<dim3(24, 10), 256, 0, stream>>>(qF, kF, vF, biasv, attnF);
    gemm_kernel<2><<<dim3(24, 8), 256, 0, stream>>>(attnF + (size_t)256 * 3072, tow, tob, hImg,
                                                    hidden, modImg + 2 * 3072, 1024, 3072, 3072);
    gemm_kernel<2><<<dim3(24, 2), 256, 0, stream>>>(attnF, taow, taob, hTxt,
                                                    enc, modTxt + 2 * 3072, 256, 3072, 3072);
    ln_mod_kernel<<<1024, 256, 0, stream>>>(hImg, modImg + 3 * 3072, modImg + 4 * 3072, xImg);
    ln_mod_kernel<<<256, 256, 0, stream>>>(hTxt, modTxt + 3 * 3072, modTxt + 4 * 3072, xTxt);
    gemm_kernel<1><<<dim3(96, 8), 256, 0, stream>>>(xImg, iw1, ib1, midImg, nullptr, nullptr, 1024, 12288, 3072);
    gemm_kernel<1><<<dim3(96, 2), 256, 0, stream>>>(xTxt, tw1, tb1, midTxt, nullptr, nullptr, 256, 12288, 3072);
    gemm_kernel<2><<<dim3(24, 8), 256, 0, stream>>>(midImg, iw2, ib2, outF + (size_t)256 * 3072,
                                                    hImg, modImg + 5 * 3072, 1024, 3072, 12288);
    gemm_kernel<2><<<dim3(24, 2), 256, 0, stream>>>(midTxt, tw2, tb2, outF,
                                                    hTxt, modTxt + 5 * 3072, 256, 3072, 12288);
}
